// DDFFModel_64295660421394
// MI455X (gfx1250) — matrix operand order repaired and checked by reading, not compiled, not run on hardware
//
#include <hip/hip_runtime.h>
#include <math.h>

// ---------------- model constants (match reference) ----------------
#define BSZ    2
#define SLQ    256
#define SLR    256
#define TT     (SLQ + SLR)      // 512
#define DM     768
#define NH     12
#define NLAYER 2
#define VOC    50257
#define FFD    3072
#define HDIM   (DM / NH)        // 64
#define NTOP   5
#define LN_EPS 1e-5f
#define NEGBIG -1e9f
#define SELRATE 0.5f
#define PARAM1 0.0f
#define PARAM2 1.0f

typedef _Float16 half16 __attribute__((ext_vector_type(16)));
typedef _Float16 half8  __attribute__((ext_vector_type(8)));
typedef float    float8 __attribute__((ext_vector_type(8)));

__device__ __forceinline__ float gelu_tanh(float x) {
    float c = 0.7978845608028654f * (x + 0.044715f * x * x * x);
    return 0.5f * x * (1.f + tanhf(c));
}

__device__ __forceinline__ half16 pack16(float4 q0, float4 q1, float4 q2, float4 q3) {
    half16 h;
    h[0]  = (_Float16)q0.x; h[1]  = (_Float16)q0.y;
    h[2]  = (_Float16)q0.z; h[3]  = (_Float16)q0.w;
    h[4]  = (_Float16)q1.x; h[5]  = (_Float16)q1.y;
    h[6]  = (_Float16)q1.z; h[7]  = (_Float16)q1.w;
    h[8]  = (_Float16)q2.x; h[9]  = (_Float16)q2.y;
    h[10] = (_Float16)q2.z; h[11] = (_Float16)q2.w;
    h[12] = (_Float16)q3.x; h[13] = (_Float16)q3.y;
    h[14] = (_Float16)q3.z; h[15] = (_Float16)q3.w;
    return h;
}

// ---------------- WMMA GEMM (f32 in): C = act(A @ B + bias) (+resid) ----
// A row-major f32, M multiple of 64. BT==0 -> B[K,N]; BT==1 -> B[N,K] (B^T).
// 4 waves/block; each wave computes a 16x32 tile (2 WMMAs share one A frag).
template <int BT>
__global__ __launch_bounds__(128)
void wmma_gemm_kernel(const float* __restrict__ A, const float* __restrict__ Bm,
                      const float* __restrict__ bias, const float* __restrict__ resid,
                      float* __restrict__ C,
                      int M, int N, int K, int act)
{
    const int lane  = threadIdx.x & 31;
    const int wave  = threadIdx.x >> 5;
    const int hl    = lane & 15;
    const int hi    = lane >> 4;
    const int tileM = blockIdx.x * 64 + wave * 16;
    const int tileN = blockIdx.y * 32;

    const int rowA  = tileM + hl;
    const int colB0 = tileN + hl;
    const int colB1 = tileN + 16 + hl;
    const int c0 = colB0 < N ? colB0 : N - 1;
    const int c1 = colB1 < N ? colB1 : N - 1;

    const float* Ap  = A + (size_t)rowA * K;
    const float* Bp0 = BT ? Bm + (size_t)c0 * K : Bm + c0;
    const float* Bp1 = BT ? Bm + (size_t)c1 * K : Bm + c1;

    float8 acc0 = {0.f, 0.f, 0.f, 0.f, 0.f, 0.f, 0.f, 0.f};
    float8 acc1 = {0.f, 0.f, 0.f, 0.f, 0.f, 0.f, 0.f, 0.f};

    for (int k0 = 0; k0 < K; k0 += 32) {
        const int abase = k0 + hi * 16;
        const float4* pa0 = (const float4*)(Ap + abase);
        const float4* pa1 = (const float4*)(Ap + abase + 8);
        const float4 x0 = pa0[0], x1 = pa0[1];
        const float4 y0 = pa1[0], y1 = pa1[1];
        half16 a;
        a[0]  = (_Float16)x0.x; a[1]  = (_Float16)x0.y;
        a[2]  = (_Float16)x0.z; a[3]  = (_Float16)x0.w;
        a[4]  = (_Float16)x1.x; a[5]  = (_Float16)x1.y;
        a[6]  = (_Float16)x1.z; a[7]  = (_Float16)x1.w;
        a[8]  = (_Float16)y0.x; a[9]  = (_Float16)y0.y;
        a[10] = (_Float16)y0.z; a[11] = (_Float16)y0.w;
        a[12] = (_Float16)y1.x; a[13] = (_Float16)y1.y;
        a[14] = (_Float16)y1.z; a[15] = (_Float16)y1.w;

        const int bbase = k0 + hi * 16;
        half16 b0, b1;
        if (BT) {
            const float4* pb0 = (const float4*)(Bp0 + bbase);
            b0 = pack16(pb0[0], pb0[1], pb0[2], pb0[3]);
            const float4* pb1 = (const float4*)(Bp1 + bbase);
            b1 = pack16(pb1[0], pb1[1], pb1[2], pb1[3]);
        } else {
#pragma unroll
            for (int i = 0; i < 16; ++i) {
                const size_t roff = (size_t)(bbase + i) * N;    // lane-coalesced
                b0[i] = (_Float16)Bp0[roff];
                b1[i] = (_Float16)Bp1[roff];
            }
        }
        acc0 = __builtin_amdgcn_wmma_f32_16x16x32_f16(
                   false, a, false, b0, (short)0, acc0, false, false);
        acc1 = __builtin_amdgcn_wmma_f32_16x16x32_f16(
                   false, a, false, b1, (short)0, acc1, false, false);
    }

    const int mbase = tileM + hi * 8;
    const int n0 = tileN + hl;
    const int n1 = n0 + 16;
    if (n0 < N) {
        const float bi = bias ? bias[n0] : 0.f;
#pragma unroll
        for (int r = 0; r < 8; ++r) {
            float v = acc0[r] + bi;
            if (act == 1) v = gelu_tanh(v);
            const size_t off = (size_t)(mbase + r) * N + n0;
            if (resid) v += resid[off];
            C[off] = v;
        }
    }
    if (n1 < N) {
        const float bi = bias ? bias[n1] : 0.f;
#pragma unroll
        for (int r = 0; r < 8; ++r) {
            float v = acc1[r] + bi;
            if (act == 1) v = gelu_tanh(v);
            const size_t off = (size_t)(mbase + r) * N + n1;
            if (resid) v += resid[off];
            C[off] = v;
        }
    }
}

// ------ WMMA GEMM (pre-converted f16): C = A_h @ B_h^T, no bias/act -----
// A_h [M,K] f16 row-major (M mult of 64); B_h [N,K] f16 row-major.
// Dominant kernel (~65% of FLOPs). Software-pipelined: next k-step's six
// b128 loads are issued before the current fragments are consumed, so each
// WMMA's operands were requested a full iteration earlier (hides L2 latency).
__global__ __launch_bounds__(128)
void wmma_gemm_f16t_kernel(const _Float16* __restrict__ A,
                           const _Float16* __restrict__ Bm,
                           float* __restrict__ C, int M, int N, int K)
{
    const int lane  = threadIdx.x & 31;
    const int wave  = threadIdx.x >> 5;
    const int hl    = lane & 15;
    const int hi    = lane >> 4;
    const int tileM = blockIdx.x * 64 + wave * 16;
    const int tileN = blockIdx.y * 32;

    const int rowA  = tileM + hl;
    const int colB0 = tileN + hl;
    const int colB1 = tileN + 16 + hl;
    const int c0 = colB0 < N ? colB0 : N - 1;
    const int c1 = colB1 < N ? colB1 : N - 1;

    const _Float16* Ap  = A + (size_t)rowA * K + hi * 16;    // a-halves base
    const _Float16* Bp0 = Bm + (size_t)c0 * K + hi * 16;    // b-halves base
    const _Float16* Bp1 = Bm + (size_t)c1 * K + hi * 16;

    float8 acc0 = {0.f, 0.f, 0.f, 0.f, 0.f, 0.f, 0.f, 0.f};
    float8 acc1 = {0.f, 0.f, 0.f, 0.f, 0.f, 0.f, 0.f, 0.f};

    // ---- prologue: fragments for k=0 ----
    half8  alo = *(const half8*)(Ap);
    half8  ahi = *(const half8*)(Ap + 8);
    half16 b0  = *(const half16*)(Bp0);
    half16 b1  = *(const half16*)(Bp1);

    // ---- steady state: load k+32 while computing k ----
#pragma unroll 2
    for (int k0 = 0; k0 < K - 32; k0 += 32) {
        const half8  alo_n = *(const half8*)(Ap + k0 + 32);
        const half8  ahi_n = *(const half8*)(Ap + k0 + 40);
        const half16 b0_n  = *(const half16*)(Bp0 + k0 + 32);
        const half16 b1_n  = *(const half16*)(Bp1 + k0 + 32);

        const half16 a = __builtin_shufflevector(alo, ahi,
            0, 1, 2, 3, 4, 5, 6, 7, 8, 9, 10, 11, 12, 13, 14, 15);
        acc0 = __builtin_amdgcn_wmma_f32_16x16x32_f16(
                   false, a, false, b0, (short)0, acc0, false, false);
        acc1 = __builtin_amdgcn_wmma_f32_16x16x32_f16(
                   false, a, false, b1, (short)0, acc1, false, false);

        alo = alo_n; ahi = ahi_n; b0 = b0_n; b1 = b1_n;
    }
    // ---- epilogue: last k-step ----
    {
        const half16 a = __builtin_shufflevector(alo, ahi,
            0, 1, 2, 3, 4, 5, 6, 7, 8, 9, 10, 11, 12, 13, 14, 15);
        acc0 = __builtin_amdgcn_wmma_f32_16x16x32_f16(
                   false, a, false, b0, (short)0, acc0, false, false);
        acc1 = __builtin_amdgcn_wmma_f32_16x16x32_f16(
                   false, a, false, b1, (short)0, acc1, false, false);
    }

    const int mbase = tileM + hi * 8;
    const int n0 = tileN + hl;
    const int n1 = n0 + 16;
    if (n0 < N) {
#pragma unroll
        for (int r = 0; r < 8; ++r)
            C[(size_t)(mbase + r) * N + n0] = acc0[r];
    }
    if (n1 < N) {
#pragma unroll
        for (int r = 0; r < 8; ++r)
            C[(size_t)(mbase + r) * N + n1] = acc1[r];
    }
}

// ---------------- f32 -> f16 conversion (grid-stride, pairs) -------------
__global__ __launch_bounds__(256)
void f32_to_f16_kernel(const float* __restrict__ src, _Float16* __restrict__ dst,
                       long n)   // n must be even (all callers: multiples of DM)
{
    long i = ((long)blockIdx.x * 256 + threadIdx.x) * 2;
    const long stride = (long)gridDim.x * 256 * 2;
    for (; i < n; i += stride) {
        const float2 v = *(const float2*)(src + i);
        dst[i]     = (_Float16)v.x;
        dst[i + 1] = (_Float16)v.y;
    }
}

// ---------------- softmax attention: one (b, h, q) row per block ---------
__global__ __launch_bounds__(128)
void attn_softmax_kernel(const float* __restrict__ Qb, const float* __restrict__ Kb,
                         const float* __restrict__ Vb, float* __restrict__ Ob,
                         int sq, int skv, int so, int Tq, int Tk,
                         int causal, float scale)
{
    extern __shared__ float smem[];
    float* s   = smem;
    float* red = smem + Tk;
    const int t = blockIdx.x, h = blockIdx.y, b = blockIdx.z;
    const int tid = threadIdx.x;
    const float* q = Qb + ((size_t)b * Tq + t) * sq + h * HDIM;
    const float* K = Kb + (size_t)b * Tk * skv + h * HDIM;
    const float* V = Vb + (size_t)b * Tk * skv + h * HDIM;
    const int kmax = causal ? (t + 1) : Tk;

    for (int j = tid; j < Tk; j += blockDim.x) {
        float acc = NEGBIG;
        if (j < kmax) {
            const float* kr = K + (size_t)j * skv;
            float d0 = 0.f;
            for (int d = 0; d < HDIM; ++d) d0 += q[d] * kr[d];
            acc = d0 * scale;
        }
        s[j] = acc;
    }
    __syncthreads();

    float m = -3.4e38f;
    for (int j = tid; j < Tk; j += blockDim.x) m = fmaxf(m, s[j]);
    red[tid] = m; __syncthreads();
    for (int st = 64; st > 0; st >>= 1) {
        if (tid < st) red[tid] = fmaxf(red[tid], red[tid + st]);
        __syncthreads();
    }
    const float mx = red[0];
    __syncthreads();

    float sum = 0.f;
    for (int j = tid; j < Tk; j += blockDim.x) {
        float e = __expf(s[j] - mx);
        s[j] = e;
        sum += e;
    }
    red[tid] = sum; __syncthreads();
    for (int st = 64; st > 0; st >>= 1) {
        if (tid < st) red[tid] += red[tid + st];
        __syncthreads();
    }
    const float inv = 1.f / red[0];

    if (tid < HDIM) {
        float acc = 0.f;
        for (int j = 0; j < Tk; ++j) acc += s[j] * V[(size_t)j * skv + tid];
        Ob[((size_t)b * Tq + t) * so + h * HDIM + tid] = acc * inv;
    }
}

// ---------------- layernorm (one row per block) --------------------------
__global__ __launch_bounds__(256)
void ln_kernel(const float* __restrict__ X, const float* __restrict__ g,
               const float* __restrict__ bt, float* __restrict__ Y, int n)
{
    __shared__ float red[256];
    const int row = blockIdx.x, tid = threadIdx.x;
    const float* x = X + (size_t)row * n;

    float s = 0.f;
    for (int i = tid; i < n; i += 256) s += x[i];
    red[tid] = s; __syncthreads();
    for (int st = 128; st > 0; st >>= 1) {
        if (tid < st) red[tid] += red[tid + st];
        __syncthreads();
    }
    const float mean = red[0] / n;
    __syncthreads();

    float v = 0.f;
    for (int i = tid; i < n; i += 256) { float d = x[i] - mean; v += d * d; }
    red[tid] = v; __syncthreads();
    for (int st = 128; st > 0; st >>= 1) {
        if (tid < st) red[tid] += red[tid + st];
        __syncthreads();
    }
    const float inv = rsqrtf(red[0] / n + LN_EPS);

    for (int i = tid; i < n; i += 256)
        Y[(size_t)row * n + i] = (x[i] - mean) * inv * g[i] + bt[i];
}

// ---------------- embedding gather ---------------------------------------
__global__ __launch_bounds__(256)
void gather_kernel(const int* __restrict__ query, const int* __restrict__ response,
                   const float* __restrict__ wte, float* __restrict__ cpri)
{
    const int bt = blockIdx.x;
    const int b = bt / TT, t = bt % TT;
    const int tok = (t < SLQ) ? query[b * SLQ + t] : response[b * SLR + (t - SLQ)];
    const float* src = wte + (size_t)tok * DM;
    for (int d = threadIdx.x; d < DM; d += 256)
        cpri[(size_t)bt * DM + d] = src[d];
}

// ---------------- top-5 of (logits1[t] + gumbel[t]) over V ---------------
__global__ __launch_bounds__(256)
void topk_kernel(const float* __restrict__ logits, const float* __restrict__ gumbel,
                 int* __restrict__ outidx)
{
    const int t = blockIdx.x, tid = threadIdx.x;
    const float* lr = logits + (size_t)t * VOC;
    const float* gr = gumbel + (size_t)t * VOC;

    float bv[NTOP]; int bix[NTOP];
#pragma unroll
    for (int k = 0; k < NTOP; ++k) { bv[k] = -3.4e38f; bix[k] = 0; }
    for (int j = tid; j < VOC; j += 256) {
        const float val = lr[j] + gr[j];
        if (val > bv[NTOP - 1]) {
            int p = NTOP - 1;
            while (p > 0 && val > bv[p - 1]) {
                bv[p] = bv[p - 1]; bix[p] = bix[p - 1]; --p;
            }
            bv[p] = val; bix[p] = j;
        }
    }
    __shared__ float sv[256 * NTOP];
    __shared__ int   si[256 * NTOP];
#pragma unroll
    for (int k = 0; k < NTOP; ++k) {
        sv[tid * NTOP + k] = bv[k];
        si[tid * NTOP + k] = bix[k];
    }
    __syncthreads();
    if (tid == 0) {
        for (int k = 0; k < NTOP; ++k) {
            float best = -3.4e38f; int bj = 0;
            for (int i = 0; i < 256 * NTOP; ++i)
                if (sv[i] > best) { best = sv[i]; bj = i; }
            outidx[t * NTOP + k] = si[bj];
            sv[bj] = -3.4e38f;
        }
    }
}

// ---------------- oracle[t] = mean_k wte[idx[t][k]] ----------------------
__global__ __launch_bounds__(256)
void oracle_kernel(const int* __restrict__ idx, const float* __restrict__ wte,
                   float* __restrict__ oracle)
{
    const int t = blockIdx.x;
    for (int d = threadIdx.x; d < DM; d += 256) {
        float s = 0.f;
#pragma unroll
        for (int k = 0; k < NTOP; ++k)
            s += wte[(size_t)idx[t * NTOP + k] * DM + d];
        oracle[(size_t)t * DM + d] = s * (1.f / NTOP);
    }
}

// ---------------- c_mod = where(select, oracle, c_pri) -------------------
__global__ __launch_bounds__(256)
void selectmix_kernel(const unsigned char* __restrict__ sel,
                      const float* __restrict__ oracle,
                      const float* __restrict__ cpri, float* __restrict__ cmod)
{
    const int bt = blockIdx.x;
    const int t = bt % TT;
    const int on = sel[bt] != 0;
    for (int d = threadIdx.x; d < DM; d += 256)
        cmod[(size_t)bt * DM + d] =
            on ? oracle[(size_t)t * DM + d] : cpri[(size_t)bt * DM + d];
}

// ---- c_output = [que_emb + 0.5*mha ; new_res]; x = P1*c_aft + P2*c_out --
__global__ __launch_bounds__(256)
void combine_kernel(const float* __restrict__ cpri, const float* __restrict__ cmod,
                    const float* __restrict__ mo, const float* __restrict__ caft,
                    float* __restrict__ coutp, float* __restrict__ x)
{
    const int bt = blockIdx.x;
    const int b = bt / TT, t = bt % TT;
    for (int d = threadIdx.x; d < DM; d += 256) {
        float co;
        if (t < SLQ)
            co = cpri[(size_t)bt * DM + d] +
                 SELRATE * mo[((size_t)b * SLQ + t) * DM + d];
        else
            co = cmod[(size_t)bt * DM + d];
        coutp[(size_t)bt * DM + d] = co;
        x[(size_t)bt * DM + d] = PARAM1 * caft[(size_t)bt * DM + d] + PARAM2 * co;
    }
}

// ---------------- latent loss: two-stage deterministic reduction ---------
__global__ __launch_bounds__(256)
void loss_part_kernel(const float* __restrict__ a, const float* __restrict__ c,
                      float* __restrict__ part, int n)
{
    __shared__ float red[256];
    float s = 0.f;
    for (int i = blockIdx.x * 256 + threadIdx.x; i < n; i += 256 * 256) {
        const float d = a[i] - c[i];
        s += d * d;
    }
    red[threadIdx.x] = s; __syncthreads();
    for (int st = 128; st > 0; st >>= 1) {
        if (threadIdx.x < st) red[threadIdx.x] += red[threadIdx.x + st];
        __syncthreads();
    }
    if (threadIdx.x == 0) part[blockIdx.x] = red[0];
}

__global__ __launch_bounds__(256)
void loss_final_kernel(const float* __restrict__ part, float* __restrict__ out)
{
    __shared__ float red[256];
    red[threadIdx.x] = part[threadIdx.x]; __syncthreads();
    for (int st = 128; st > 0; st >>= 1) {
        if (threadIdx.x < st) red[threadIdx.x] += red[threadIdx.x + st];
        __syncthreads();
    }
    if (threadIdx.x == 0) out[0] = red[0];
}

// ======================= host side ======================================
struct GptW {
    const float *wqkv, *bqkv, *wo, *bo, *ln1g, *ln1b, *ln2g, *ln2b;
    const float *w1, *b1, *w2, *b2, *lnfg, *lnfb;
};

static inline void gemm(hipStream_t s, const float* A, const float* Bm,
                        const float* bias, const float* resid, float* C,
                        int M, int N, int K, int bT, int act)
{
    dim3 grid((M + 63) / 64, (N + 31) / 32);
    if (bT)
        wmma_gemm_kernel<1><<<grid, 128, 0, s>>>(A, Bm, bias, resid, C, M, N, K, act);
    else
        wmma_gemm_kernel<0><<<grid, 128, 0, s>>>(A, Bm, bias, resid, C, M, N, K, act);
}

static inline void gemm_f16t(hipStream_t s, const _Float16* A, const _Float16* Bm,
                             float* C, int M, int N, int K)
{
    dim3 grid((M + 63) / 64, (N + 31) / 32);
    wmma_gemm_f16t_kernel<<<grid, 128, 0, s>>>(A, Bm, C, M, N, K);
}

static void gpt_forward(hipStream_t s, const GptW& W, float* x, float* xln,
                        float* qkv, float* atty, float* ffh)
{
    const int M = BSZ * TT;
    const float scale = 0.125f;  // 1/sqrt(64)
    for (int l = 0; l < NLAYER; ++l) {
        ln_kernel<<<M, 256, 0, s>>>(x, W.ln1g + l * DM, W.ln1b + l * DM, xln, DM);
        gemm(s, xln, W.wqkv + (size_t)l * DM * 3 * DM, W.bqkv + l * 3 * DM,
             nullptr, qkv, M, 3 * DM, DM, 0, 0);
        dim3 ag(TT, NH, BSZ);
        attn_softmax_kernel<<<ag, 128, (TT + 128) * sizeof(float), s>>>(
            qkv, qkv + DM, qkv + 2 * DM, atty, 3 * DM, 3 * DM, DM, TT, TT, 1, scale);
        gemm(s, atty, W.wo + (size_t)l * DM * DM, W.bo + l * DM, x, x, M, DM, DM, 0, 0);
        ln_kernel<<<M, 256, 0, s>>>(x, W.ln2g + l * DM, W.ln2b + l * DM, xln, DM);
        gemm(s, xln, W.w1 + (size_t)l * DM * FFD, W.b1 + l * FFD, nullptr, ffh,
             M, FFD, DM, 0, 1);
        gemm(s, ffh, W.w2 + (size_t)l * FFD * DM, W.b2 + l * DM, x, x, M, DM, FFD, 0, 0);
    }
    ln_kernel<<<M, 256, 0, s>>>(x, W.lnfg, W.lnfb, xln, DM);
}

extern "C" void kernel_launch(void* const* d_in, const int* in_sizes, int n_in,
                              void* d_out, int out_size, void* d_ws, size_t ws_size,
                              hipStream_t stream)
{
    (void)in_sizes; (void)n_in; (void)out_size; (void)ws_size;
    // -------- inputs (setup_inputs order) --------
    const int*   query      = (const int*)d_in[0];
    const int*   response   = (const int*)d_in[1];
    const float* gumbel     = (const float*)d_in[2];
    const unsigned char* sel = (const unsigned char*)d_in[3];   // jax bool -> 1 byte
    const float* wte        = (const float*)d_in[4];
    GptW W;
    W.wqkv = (const float*)d_in[5];  W.bqkv = (const float*)d_in[6];
    W.wo   = (const float*)d_in[7];  W.bo   = (const float*)d_in[8];
    W.ln1g = (const float*)d_in[9];  W.ln1b = (const float*)d_in[10];
    W.ln2g = (const float*)d_in[11]; W.ln2b = (const float*)d_in[12];
    W.w1   = (const float*)d_in[13]; W.b1   = (const float*)d_in[14];
    W.w2   = (const float*)d_in[15]; W.b2   = (const float*)d_in[16];
    W.lnfg = (const float*)d_in[17]; W.lnfb = (const float*)d_in[18];
    const float* attn_in_w  = (const float*)d_in[19];
    const float* attn_in_b  = (const float*)d_in[20];
    const float* attn_out_w = (const float*)d_in[21];
    const float* attn_out_b = (const float*)d_in[22];
    const float* ll_w       = (const float*)d_in[23];
    const float* ll_b       = (const float*)d_in[24];

    float* out = (float*)d_out;                      // B*T*V logits + 1 loss

    // -------- workspace carve (64B-aligned slabs) --------
    float* ws = (float*)d_ws;
    size_t o = 0;
    auto alloc = [&](size_t n) {                 // n in floats
        o = (o + 15) & ~(size_t)15;              // 64B alignment
        float* p = ws + o; o += n; return p;
    };
    float* x      = alloc((size_t)BSZ * TT * DM);
    float* xln    = alloc((size_t)BSZ * TT * DM);
    float* qkv    = alloc((size_t)BSZ * TT * 3 * DM);
    float* atty   = alloc((size_t)BSZ * TT * DM);
    float* ffh    = alloc((size_t)BSZ * TT * FFD);
    float* cpri   = alloc((size_t)BSZ * TT * DM);
    float* cmod   = alloc((size_t)BSZ * TT * DM);
    float* coutp  = alloc((size_t)BSZ * TT * DM);
    float* caft   = alloc((size_t)BSZ * TT * DM);
    float* oracle = alloc((size_t)TT * DM);
    float* mq     = alloc((size_t)BSZ * SLQ * DM);
    float* mk     = alloc((size_t)BSZ * SLR * DM);
    float* mv     = alloc((size_t)BSZ * SLR * DM);
    float* my     = alloc((size_t)BSZ * SLQ * DM);
    float* mo     = alloc((size_t)BSZ * SLQ * DM);
    float* lpart  = alloc(256);
    float* logits1 = alloc((size_t)TT * VOC);
    _Float16* wteh = (_Float16*)alloc(((size_t)VOC * DM + 1) / 2);  // f16 wte
    _Float16* xh   = (_Float16*)alloc(((size_t)BSZ * TT * DM + 1) / 2);
    int*      topidx = (int*)alloc((size_t)TT * NTOP);

    const int M  = BSZ * TT;
    const int ND = BSZ * TT * DM;

    // 0) one-time f16 conversion of wte (reused by the logits GEMMs)
    f32_to_f16_kernel<<<4096, 256, 0, stream>>>(wte, wteh, (long)VOC * DM);

    // 1) embeddings: c_pri = wte[concat(query, response)]
    gather_kernel<<<M, 256, 0, stream>>>(query, response, wte, cpri);
    hipMemcpyAsync(x, cpri, (size_t)ND * sizeof(float),
                   hipMemcpyDeviceToDevice, stream);

    // 2) GPT pass 1
    gpt_forward(stream, W, x, xln, qkv, atty, ffh);

    // 3) logits for last batch only: out1[-1] = lnf(x)[B-1] @ wte^T   (f16 path)
    f32_to_f16_kernel<<<512, 256, 0, stream>>>(
        xln + (size_t)(BSZ - 1) * TT * DM, xh, (long)TT * DM);
    gemm_f16t(stream, xh, wteh, logits1, TT, VOC, DM);

    // 4) top-5 of (logits + gumbel) (softmax & /TAU are monotone) -> oracle
    topk_kernel<<<TT, 256, 0, stream>>>(logits1, gumbel, topidx);
    oracle_kernel<<<TT, 256, 0, stream>>>(topidx, wte, oracle);

    // 5) c_mod = where(select, oracle, c_pri)
    selectmix_kernel<<<M, 256, 0, stream>>>(sel, oracle, cpri, cmod);

    // 6) MHA(que_emb, new_res): q/k/v GEMMs per batch (strided slices)
    for (int b = 0; b < BSZ; ++b) {
        const float* tgt = cpri + (size_t)b * TT * DM;                    // que_emb
        const float* mem = cmod + (size_t)b * TT * DM + (size_t)SLQ * DM; // new_res
        gemm(stream, tgt, attn_in_w,                       attn_in_b,
             nullptr, mq + (size_t)b * SLQ * DM, SLQ, DM, DM, 1, 0);
        gemm(stream, mem, attn_in_w + (size_t)DM * DM,     attn_in_b + DM,
             nullptr, mk + (size_t)b * SLR * DM, SLR, DM, DM, 1, 0);
        gemm(stream, mem, attn_in_w + (size_t)2 * DM * DM, attn_in_b + 2 * DM,
             nullptr, mv + (size_t)b * SLR * DM, SLR, DM, DM, 1, 0);
    }
    {
        dim3 ag(SLQ, NH, BSZ);
        attn_softmax_kernel<<<ag, 128, (SLR + 128) * sizeof(float), stream>>>(
            mq, mk, mv, my, DM, DM, DM, SLQ, SLR, 0, 0.125f);
    }
    gemm(stream, my, attn_out_w, attn_out_b, nullptr, mo, BSZ * SLQ, DM, DM, 1, 0);

    // 7) c_aft = c_mod @ ll_w^T + ll_b
    gemm(stream, cmod, ll_w, ll_b, nullptr, caft, M, DM, DM, 1, 0);

    // 8) c_output, c_multi (-> x)
    combine_kernel<<<M, 256, 0, stream>>>(cpri, cmod, mo, caft, coutp, x);

    // 9) latent loss (deterministic two-stage reduction)
    loss_part_kernel<<<256, 256, 0, stream>>>(caft, coutp, lpart, ND);
    loss_final_kernel<<<1, 256, 0, stream>>>(lpart, out + (size_t)M * VOC);

    // 10) GPT pass 2 + full logits straight into d_out (f16 path)
    gpt_forward(stream, W, x, xln, qkv, atty, ffh);
    f32_to_f16_kernel<<<1024, 256, 0, stream>>>(xln, xh, (long)M * DM);
    gemm_f16t(stream, xh, wteh, out, M, VOC, DM);
}
